// GAT_26757646254515
// MI455X (gfx1250) — compile-verified, matrix operand order checked by reading, not run on hardware
//
#include <hip/hip_runtime.h>
#include <hip/hip_bf16.h>

typedef __bf16 bf16;
typedef __attribute__((ext_vector_type(16))) __bf16 v16bf;
typedef __attribute__((ext_vector_type(8)))  __bf16 v8bf;
typedef __attribute__((ext_vector_type(8)))  float  v8f;

#define GN 50000
#define GE 800000
#define GF 64
#define GH 4
#define HF 256            // H*F
#define GT (GE + GN)      // edges + self loops
#define MTILES 3125       // GN / 16

// ---------------------------------------------------------------- utilities

__global__ void fill_kernel(float* __restrict__ p, float v, int n) {
  int i = blockIdx.x * blockDim.x + threadIdx.x;
  if (i < n) p[i] = v;
}

__global__ void to_bf16_kernel(const float* __restrict__ x, bf16* __restrict__ y, int n) {
  int i = blockIdx.x * blockDim.x + threadIdx.x;
  if (i < n) y[i] = (bf16)x[i];
}

__device__ inline void atomicMaxF(float* addr, float val) {
  unsigned int* ua = (unsigned int*)addr;
  unsigned int old = __float_as_uint(*addr);
  while (__uint_as_float(old) < val) {
    unsigned int assumed = old;
    old = atomicCAS(ua, assumed, __float_as_uint(val));
    if (old == assumed) break;
  }
}

// ------------------------------------------------ setup: self loops, weights

__global__ void deg_kernel(const long long* __restrict__ dst,
                           const float* __restrict__ eattr,
                           float* __restrict__ deg, float* __restrict__ lattr) {
  int e = blockIdx.x * blockDim.x + threadIdx.x;
  if (e >= GE) return;
  long long d = dst[e];
  atomicAdd(&deg[d], 1.0f);
  atomicAdd(&lattr[d], eattr[e]);
}

__global__ void loopdiv_kernel(float* __restrict__ lattr, const float* __restrict__ deg) {
  int i = blockIdx.x * blockDim.x + threadIdx.x;
  if (i < GN) lattr[i] /= fmaxf(deg[i], 1.0f);
}

// Transposed bf16 weights: Wt[l][n*64+k] = W_l[k*256+n], l = 2*pass + side
__global__ void prep_w_kernel(const float* __restrict__ glw,
                              const float* __restrict__ grw,
                              bf16* __restrict__ Wt) {
  int idx = blockIdx.x * blockDim.x + threadIdx.x;
  if (idx >= 4 * HF * GF) return;
  int l = idx / (HF * GF);
  int rem = idx % (HF * GF);
  int n = rem / GF, k = rem % GF;
  int p = l >> 1;
  const float* W = (l & 1) ? grw : glw;
  Wt[idx] = (bf16)W[(size_t)p * GF * HF + (size_t)k * HF + n];
}

// ------------------------------------------------------------ WMMA dual GEMM
// One launch computes  YL = X@Wl + bl  and  YR = X@Wr + br.
// Block = 8 waves; both Wt matrices (2 x 32KB bf16) + biases staged in LDS.
// Each wave owns one 16-row M-strip: A fragments loaded ONCE from global,
// then sweeps 16 N-tiles x 2 matrices with B fragments from ds_load.
// K = 64 -> 2 x v_wmma_f32_16x16x32_bf16 per tile; 64 WMMAs per wave total.

__device__ inline v16bf load_frag(const bf16* __restrict__ rowbase, int kb, int koff) {
  // element e -> K = koff + 16*(e>>3) + kb + (e&7)
  v8bf lo = *reinterpret_cast<const v8bf*>(rowbase + koff + kb);
  v8bf hi = *reinterpret_cast<const v8bf*>(rowbase + koff + kb + 16);
  v16bf r;
#pragma unroll
  for (int i = 0; i < 8; ++i) { r[i] = lo[i]; r[8 + i] = hi[i]; }
  return r;
}

__global__ void __launch_bounds__(256)
gemm_dual_wmma_kernel(const bf16* __restrict__ X,
                      const bf16* __restrict__ WtL, const bf16* __restrict__ WtR,
                      const float* __restrict__ biasL, const float* __restrict__ biasR,
                      float* __restrict__ YL, float* __restrict__ YR) {
  __shared__ bf16  sW[2][HF * GF];   // 2 * 16384 bf16 = 64 KB
  __shared__ float sB[2][HF];        // 2 KB

  int t = threadIdx.x;
  // cooperative stage: 2048 x 16B chunks per matrix, 256 threads -> 8 iters
  const uint4* gL = (const uint4*)WtL;
  const uint4* gR = (const uint4*)WtR;
  uint4* sL = (uint4*)&sW[0][0];
  uint4* sR = (uint4*)&sW[1][0];
#pragma unroll
  for (int i = 0; i < 8; ++i) {
    sL[t + i * 256] = gL[t + i * 256];
    sR[t + i * 256] = gR[t + i * 256];
  }
  if (t < HF) { sB[0][t] = biasL[t]; sB[1][t] = biasR[t]; }
  __syncthreads();

  int wid  = t >> 5;
  int lane = t & 31;
  int mt = blockIdx.x * 8 + wid;
  if (mt >= MTILES) return;

  int row = mt * 16 + (lane & 15);
  int kb = (lane >> 4) << 3;          // 0 or 8
  const bf16* pA = X + (size_t)row * GF;
  v16bf a0 = load_frag(pA, kb, 0);
  v16bf a1 = load_frag(pA, kb, 32);
  int rbase = mt * 16 + ((lane >> 4) << 3);

#pragma unroll
  for (int m = 0; m < 2; ++m) {
    float* __restrict__ Y = m ? YR : YL;
#pragma unroll
    for (int nt = 0; nt < 16; ++nt) {
      int ncol = nt * 16 + (lane & 15);
      const bf16* pB = &sW[m][(size_t)ncol * GF];
      v16bf b0 = load_frag(pB, kb, 0);
      v16bf b1 = load_frag(pB, kb, 32);
      v8f c = {};
      c = __builtin_amdgcn_wmma_f32_16x16x32_bf16(false, a0, false, b0, (short)0, c, false, false);
      c = __builtin_amdgcn_wmma_f32_16x16x32_bf16(false, a1, false, b1, (short)0, c, false, false);
      float bv = sB[m][ncol];
#pragma unroll
      for (int r = 0; r < 8; ++r)
        Y[(size_t)(rbase + r) * HF + ncol] = c[r] + bv;
    }
  }
}

// --------------------------------------------------------------- edge phase
// One wave per edge; lane handles 8 consecutive (h,f) slots -> head = lane>>3.

__global__ void __launch_bounds__(256)
edge_logits_kernel(const long long* __restrict__ src, const long long* __restrict__ dst,
                   const float* __restrict__ eattr, const float* __restrict__ lattr,
                   const float* __restrict__ xl, const float* __restrict__ xr,
                   const float* __restrict__ wev, const float* __restrict__ att,
                   float* __restrict__ logits, float* __restrict__ lmax) {
  int gw = (blockIdx.x * 256 + threadIdx.x) >> 5;
  int lane = threadIdx.x & 31;
  if (gw >= GT) return;
  long long s, d; float ea;
  if (gw < GE) { s = src[gw]; d = dst[gw]; ea = eattr[gw]; }
  else         { s = d = gw - GE; ea = lattr[gw - GE]; }

  const float* pl = xl + (size_t)s * HF + lane * 8;
  const float* pr = xr + (size_t)d * HF + lane * 8;
  float partial = 0.f;
#pragma unroll
  for (int i = 0; i < 8; ++i) {
    int hf = lane * 8 + i;
    float m = pl[i] + pr[i] + ea * wev[hf];
    m = m > 0.f ? m : 0.2f * m;                  // leaky_relu 0.2
    partial += m * att[hf];
  }
  partial += __shfl_xor(partial, 1);
  partial += __shfl_xor(partial, 2);
  partial += __shfl_xor(partial, 4);
  if ((lane & 7) == 0) {
    int h = lane >> 3;
    logits[(size_t)gw * GH + h] = partial;
    atomicMaxF(&lmax[(size_t)d * GH + h], partial);
  }
}

__global__ void edge_exp_kernel(const long long* __restrict__ dst,
                                float* __restrict__ logits,
                                const float* __restrict__ lmax,
                                float* __restrict__ denom) {
  int idx = blockIdx.x * blockDim.x + threadIdx.x;
  if (idx >= GT * GH) return;
  int e = idx >> 2, h = idx & 3;
  long long d = (e < GE) ? dst[e] : (long long)(e - GE);
  float p = __expf(logits[idx] - lmax[(size_t)d * GH + h]);
  logits[idx] = p;
  atomicAdd(&denom[(size_t)d * GH + h], p);
}

__global__ void __launch_bounds__(256)
edge_scatter_kernel(const long long* __restrict__ src, const long long* __restrict__ dst,
                    const float* __restrict__ xl, const float* __restrict__ plog,
                    const float* __restrict__ denom, float* __restrict__ out) {
  int gw = (blockIdx.x * 256 + threadIdx.x) >> 5;
  int lane = threadIdx.x & 31;
  if (gw >= GT) return;
  long long s, d;
  if (gw < GE) { s = src[gw]; d = dst[gw]; }
  else         { s = d = gw - GE; }
  int h = lane >> 3;
  float alpha = plog[(size_t)gw * GH + h] / (denom[(size_t)d * GH + h] + 1e-16f);
  const float* pl = xl + (size_t)s * HF + lane * 8;
  float* po = out + (size_t)d * HF + lane * 8;
#pragma unroll
  for (int i = 0; i < 8; ++i) atomicAdd(&po[i], alpha * pl[i]);
}

// ------------------------------------------------ node update + mean pooling
// 256 threads = 4 nodes x 64 features; LDS reduce before global mean atomics.

__global__ void __launch_bounds__(256)
node_update_kernel(const float* __restrict__ out, const float* __restrict__ gbias,
                   float* __restrict__ xcur, float* __restrict__ gsum) {
  __shared__ float sred[GF];
  int t = threadIdx.x;
  int f = t & 63;
  int node = blockIdx.x * 4 + (t >> 6);
  if (t < GF) sred[t] = 0.f;
  __syncthreads();
  float acc = 0.f;
#pragma unroll
  for (int h = 0; h < GH; ++h) acc += out[(size_t)node * HF + h * GF + f];
  acc = acc * 0.25f + gbias[f];                  // head mean + bias
  float v = acc > 0.f ? acc : 0.01f * acc;       // leaky_relu 0.01
  float xn = xcur[(size_t)node * GF + f] + v;    // residual
  xcur[(size_t)node * GF + f] = xn;
  atomicAdd(&sred[f], xn);
  __syncthreads();
  if (t < GF) atomicAdd(&gsum[t], sred[t]);
}

__global__ void __launch_bounds__(256)
mean_accum_kernel(const float* __restrict__ x, float* __restrict__ gsum) {
  __shared__ float sred[GF];
  int t = threadIdx.x;
  int f = t & 63;
  int node = blockIdx.x * 4 + (t >> 6);
  if (t < GF) sred[t] = 0.f;
  __syncthreads();
  atomicAdd(&sred[f], x[(size_t)node * GF + f]);
  __syncthreads();
  if (t < GF) atomicAdd(&gsum[t], sred[t]);
}

// ------------------------------------------------------------------ MLP head

__global__ void __launch_bounds__(256)
mlp_kernel(const float* __restrict__ gsum, const float* __restrict__ ptype,
           const float* __restrict__ W1, const float* __restrict__ b1,
           const float* __restrict__ W2, const float* __restrict__ b2,
           const float* __restrict__ W3, const float* __restrict__ b3,
           float* __restrict__ outp) {
  __shared__ float g[193];
  __shared__ float h1[96];
  __shared__ float h2[96];
  int t = threadIdx.x;
  if (t < 192) g[t] = gsum[t] * (1.0f / (float)GN);
  if (t == 192) g[192] = ptype[0];
  __syncthreads();
  if (t < 96) {
    float a = b1[t];
    for (int k = 0; k < 193; ++k) a += g[k] * W1[k * 96 + t];
    h1[t] = a > 0.f ? a : 0.01f * a;
  }
  __syncthreads();
  if (t < 96) {
    float a = b2[t];
    for (int k = 0; k < 96; ++k) a += h1[k] * W2[k * 96 + t];
    h2[t] = a > 0.f ? a : 0.01f * a;
  }
  __syncthreads();
  if (t < 10) {
    float a = b3[t];
    for (int k = 0; k < 96; ++k) a += h2[k] * W3[k * 10 + t];
    outp[t] = a;
  }
}

// ------------------------------------------------------------------- launch

extern "C" void kernel_launch(void* const* d_in, const int* in_sizes, int n_in,
                              void* d_out, int out_size, void* d_ws, size_t ws_size,
                              hipStream_t stream) {
  const float*     x     = (const float*)d_in[0];
  const long long* eidx  = (const long long*)d_in[1];
  const float*     eattr = (const float*)d_in[2];
  const float*     ptype = (const float*)d_in[3];
  const float*     glw   = (const float*)d_in[5];
  const float*     glb   = (const float*)d_in[6];
  const float*     grw   = (const float*)d_in[7];
  const float*     grb   = (const float*)d_in[8];
  const float*     gew   = (const float*)d_in[9];
  const float*     gatt  = (const float*)d_in[10];
  const float*     gbias = (const float*)d_in[11];
  const float*     W1    = (const float*)d_in[12];
  const float*     b1    = (const float*)d_in[13];
  const float*     W2    = (const float*)d_in[14];
  const float*     b2    = (const float*)d_in[15];
  const float*     W3    = (const float*)d_in[16];
  const float*     b3    = (const float*)d_in[17];

  const long long* srcp = eidx;
  const long long* dstp = eidx + GE;

  // workspace layout (floats)
  float* ws = (float*)d_ws;
  float* xl     = ws;                      // N*256
  float* xrout  = xl     + (size_t)GN*HF;  // N*256 (xr, then reused as out accum)
  float* logits = xrout  + (size_t)GN*HF;  // (E+N)*4
  float* lmax   = logits + (size_t)GT*GH;  // N*4
  float* denom  = lmax   + (size_t)GN*GH;  // N*4
  float* deg    = denom  + (size_t)GN*GH;  // N
  float* lattr  = deg    + GN;             // N
  float* xcur   = lattr  + GN;             // N*64
  float* gsum   = xcur   + (size_t)GN*GF;  // 256 (uses 192)
  bf16*  xbf    = (bf16*)(gsum + 256);     // N*64 bf16
  bf16*  Wt     = xbf + (size_t)GN*GF;     // 4*256*64 bf16

  const int TB = 256;
  auto cdiv = [](long long a, long long b) { return (int)((a + b - 1) / b); };

  // ---- one-time setup
  fill_kernel<<<cdiv(GN, TB), TB, 0, stream>>>(deg, 0.f, GN);
  fill_kernel<<<cdiv(GN, TB), TB, 0, stream>>>(lattr, 0.f, GN);
  fill_kernel<<<1, TB, 0, stream>>>(gsum, 0.f, 256);
  deg_kernel<<<cdiv(GE, TB), TB, 0, stream>>>(dstp, eattr, deg, lattr);
  loopdiv_kernel<<<cdiv(GN, TB), TB, 0, stream>>>(lattr, deg);
  prep_w_kernel<<<cdiv(4 * HF * GF, TB), TB, 0, stream>>>(glw, grw, Wt);
  hipMemcpyAsync(xcur, x, (size_t)GN * GF * sizeof(float),
                 hipMemcpyDeviceToDevice, stream);
  mean_accum_kernel<<<GN / 4, TB, 0, stream>>>(xcur, gsum + 0);

  // ---- GATv2 passes
  for (int p = 0; p < 2; ++p) {
    to_bf16_kernel<<<cdiv((long long)GN * GF, TB), TB, 0, stream>>>(xcur, xbf, GN * GF);

    // fused xl/xr transform: A read once, weights from LDS, 64 WMMAs/wave
    gemm_dual_wmma_kernel<<<cdiv(MTILES, 8), TB, 0, stream>>>(
        xbf,
        Wt + (size_t)(2 * p) * HF * GF, Wt + (size_t)(2 * p + 1) * HF * GF,
        glb + p * HF, grb + p * HF,
        xl, xrout);

    fill_kernel<<<cdiv(GN * GH, TB), TB, 0, stream>>>(lmax, -3.4e38f, GN * GH);
    fill_kernel<<<cdiv(GN * GH, TB), TB, 0, stream>>>(denom, 0.f, GN * GH);

    edge_logits_kernel<<<cdiv((long long)GT * 32, TB), TB, 0, stream>>>(
        srcp, dstp, eattr, lattr, xl, xrout, gew + p * HF, gatt + p * HF,
        logits, lmax);
    edge_exp_kernel<<<cdiv((long long)GT * GH, TB), TB, 0, stream>>>(
        dstp, logits, lmax, denom);

    // xr no longer needed; reuse as the output accumulator
    fill_kernel<<<cdiv((long long)GN * HF, TB), TB, 0, stream>>>(xrout, 0.f, GN * HF);
    edge_scatter_kernel<<<cdiv((long long)GT * 32, TB), TB, 0, stream>>>(
        srcp, dstp, xl, logits, denom, xrout);

    node_update_kernel<<<GN / 4, TB, 0, stream>>>(
        xrout, gbias + p * GF, xcur, gsum + (p + 1) * GF);
  }

  // ---- pooled MLP head -> 10 outputs
  mlp_kernel<<<1, TB, 0, stream>>>(gsum, ptype, W1, b1, W2, b2, W3, b3,
                                   (float*)d_out);
}
